// GCNBlock_69063074119738
// MI455X (gfx1250) — matrix operand order repaired and checked by reading, not compiled, not run on hardware
//
#include <hip/hip_runtime.h>
#include <hip/hip_bf16.h>
#include <math.h>

// ---------------------------------------------------------------------------
// GCN block for MI455X (gfx1250, wave32, WMMA).
//
// Pipeline:
//   1. stats_kernel      : per-(patch m, channel c) AdaIN affine (scale, shift)
//   2. assemble_hr       : gather-style kNN + AdaIN + overlap-average -> knn_hr
//                          (f32 for final blend, f16 for conv input)
//   3. cvt_f32_f16       : conv weights -> f16 (natural OIHW flat == [n][k])
//   4. conv_wmma<5x5,s2> : implicit-GEMM conv + lrelu (v_wmma_f32_16x16x32_f16,
//                          one wave = 16x64 output strip = 4 WMMA per K-step)
//   5. conv_wmma<3x3>    : 3x3 conv + lrelu
//   6. conv_wmma<3x3>    : 3x3 conv (no act) -> klr (f32)
//   7. weight_logits x2  : weightnet (1x1 convs, LDS-staged weights) -> logits
//   8. combine x2        : softmax over K + weighted sum -> d_out
// ---------------------------------------------------------------------------

typedef __attribute__((ext_vector_type(16))) _Float16 v16h;
typedef __attribute__((ext_vector_type(8)))  float    v8f;

#define K_NN   5
#define C0_    64
#define DN_    32
#define HH     64
#define WW     64
#define HS     128
#define WSZ    128
#define MG     62            // query patch grid (62x62 = 3844)
#define NG     30            // database patch grid (30x30 = 900)
#define MPATCH (MG*MG)
#define NEGS   0.2f
#define EPSV   1e-5f

// ---------------------------------------------------------------------------
// 1. AdaIN stats: per (m, c) compute scale = ystd/zstd, shift = ymean - zmean*scale
// ---------------------------------------------------------------------------
__global__ void __launch_bounds__(256)
stats_kernel(const float* __restrict__ y, const int* __restrict__ idxk,
             float* __restrict__ scale, float* __restrict__ shift)
{
  int t = blockIdx.x * blockDim.x + threadIdx.x;
  if (t >= MPATCH * C0_) return;
  int mp = t >> 6;
  int c  = t & 63;
  int a  = mp / MG, bq = mp - a * MG;
  const float* yc = y + (size_t)c * HH * WW;

  float s = 0.f, s2 = 0.f;
#pragma unroll
  for (int p1 = 0; p1 < 3; ++p1)
#pragma unroll
    for (int p2 = 0; p2 < 3; ++p2) {
      float v = yc[(a + p1) * WW + bq + p2];
      s += v; s2 += v * v;
    }
  float ym   = s * (1.f / 9.f);
  float yvar = (s2 - 9.f * ym * ym) * (1.f / 8.f);
  float ystd = sqrtf(fmaxf(yvar, 0.f) + EPSV);

  float zs = 0.f, zs2 = 0.f;
  for (int k = 0; k < K_NN; ++k) {
    int idx = idxk[mp * K_NN + k];
    int n1 = idx / NG, n2 = idx - n1 * NG;
    const float* base = yc + (2 * n1) * WW + 2 * n2;   // 6x6 block in y
#pragma unroll
    for (int i = 0; i < 6; ++i)
#pragma unroll
      for (int j = 0; j < 6; ++j) {
        float v = base[i * WW + j];
        zs += v; zs2 += v * v;
      }
  }
  float zm   = zs * (1.f / 180.f);
  float zvar = (zs2 - 180.f * zm * zm) * (1.f / 179.f);
  float zstd = sqrtf(fmaxf(zvar, 0.f) + EPSV);

  float sc = ystd / zstd;
  scale[t] = sc;
  shift[t] = ym - zm * sc;
}

// ---------------------------------------------------------------------------
// 2. Gather-style reassembly: output (k,c,Y,X) averages <=9 overlapping
//    patch contributions, each an affine-transformed gather from y.
// ---------------------------------------------------------------------------
__global__ void __launch_bounds__(256)
assemble_hr(const float* __restrict__ y, const int* __restrict__ idxk,
            const float* __restrict__ scale, const float* __restrict__ shift,
            float* __restrict__ khr, _Float16* __restrict__ khrh)
{
  int t = blockIdx.x * blockDim.x + threadIdx.x;
  if (t >= K_NN * C0_ * HS * WSZ) return;
  int X = t & (WSZ - 1);
  int Y = (t >> 7) & (HS - 1);
  int c = (t >> 14) & 63;
  int k = t >> 20;
  const float* yc = y + (size_t)c * HH * WW;

  int aLo = (Y - 4) > 0 ? (Y - 4) >> 1 : 0;
  int aHi = (Y >> 1) < (MG - 1) ? (Y >> 1) : (MG - 1);
  int bLo = (X - 4) > 0 ? (X - 4) >> 1 : 0;
  int bHi = (X >> 1) < (MG - 1) ? (X >> 1) : (MG - 1);

  float sum = 0.f, cnt = 0.f;
  for (int a = aLo; a <= aHi; ++a) {
    int i = Y - 2 * a;                 // 0..5
    int p1 = i >> 1, s1 = i & 1;
    for (int bb = bLo; bb <= bHi; ++bb) {
      int j = X - 2 * bb;
      int p2 = j >> 1, s2 = j & 1;
      int mp  = a * MG + bb;
      int idx = idxk[mp * K_NN + k];
      int n1 = idx / NG, n2 = idx - n1 * NG;
      float raw = yc[(2 * (n1 + p1) + s1) * WW + 2 * (n2 + p2) + s2];
      int mc = mp * C0_ + c;
      sum += raw * scale[mc] + shift[mc];
      cnt += 1.f;
    }
  }
  float v = sum / cnt;
  khr[t]  = v;
  khrh[t] = (_Float16)v;
}

// ---------------------------------------------------------------------------
// 3. f32 -> f16 elementwise (weight conversion; OIHW flat == [n][k] n-major)
// ---------------------------------------------------------------------------
__global__ void __launch_bounds__(256)
cvt_f32_f16(const float* __restrict__ in, _Float16* __restrict__ out, int n)
{
  int t = blockIdx.x * blockDim.x + threadIdx.x;
  if (t < n) out[t] = (_Float16)in[t];
}

// ---------------------------------------------------------------------------
// 4-6. Implicit-GEMM conv with V_WMMA_F32_16X16X32_F16.
// One wave = one 16(M pixels) x 64(N all outch) strip: 4 accumulators share
// each A-tile (amortizes the scattered im2col gathers 4x). K in steps of 32.
// All dims compile-time -> no integer division in the hot path.
//
// Layouts as used here (wave32; C/D per CDNA5 ISA §7.12.2):
//   A (16x32 f16): lane<16 holds row M=lane, K {kb+0..15} in V0..7;
//                  lane>=16 holds K {kb+16..31} (contiguous K, same order as B).
//   B (32x16 f16): row striped across lanes within one VGPR -> lane = N (mod 16);
//                  lanes 0-15 carry K {kb+0..15}, lanes 16-31 carry K {kb+16..31}.
//                  With n-major f16 weights W[n][k] a lane's 16 K-values are
//                  contiguous -> two uint4 loads.
//   C/D (16x16 f32): lane gives N=lane&15; VGPR r -> M = (lane<16 ? r : 8+r).
// ---------------------------------------------------------------------------
union AB16 { v16h v; uint4 q[2]; _Float16 h[16]; };

template<int KHW, int KWW, int STRIDE, int PAD, int Hin, int Win>
__device__ __forceinline__ _Float16
im2col_fetch(const _Float16* __restrict__ inimg, int kk, int oy, int ox)
{
  int ci = kk / KHW;
  int p  = kk - ci * KHW;
  int dy = p / KWW;
  int dx = p - dy * KWW;
  int iy = oy * STRIDE + dy - PAD;
  int ix = ox * STRIDE + dx - PAD;
  _Float16 r = (_Float16)0.f;
  if (iy >= 0 && iy < Hin && ix >= 0 && ix < Win)
    r = inimg[(ci * Hin + iy) * Win + ix];
  return r;
}

template<int KH, int KWW, int STRIDE, int PAD, bool LRELU,
         int Hin, int Win, int Hout, int Wout>
__global__ void __launch_bounds__(128)
conv_wmma(const _Float16* __restrict__ inh,   // (5, 64, Hin, Win) f16
          const _Float16* __restrict__ wh,    // (64, 64*KH*KW) f16, n-major
          const float* __restrict__ bias,     // (64)
          float* __restrict__ outf,           // (5, 64, Hout, Wout) or null
          _Float16* __restrict__ outh)        // same, f16, or null
{
  constexpr int KHW    = KH * KWW;
  constexpr int Ktot   = 64 * KHW;
  constexpr int MTILES = (Hout * Wout) / 16;

  int lane  = threadIdx.x & 31;
  int wave  = blockIdx.x * 4 + (threadIdx.x >> 5);
  int img   = wave / MTILES;
  int mtile = wave - img * MTILES;

  int  row   = lane & 15;
  int  m     = mtile * 16 + row;          // output pixel (A row)
  int  oy    = m / Wout, ox = m % Wout;   // compile-time Wout -> shift/mask
  bool lo    = lane < 16;
  int  koffA = lo ? 0 : 8;
  int  koffB = lo ? 0 : 16;

  const _Float16* inimg = inh + (size_t)img * 64 * Hin * Win;
  const _Float16* wrow  = wh + (size_t)row * Ktot;   // ntile 0; +16*Ktot per tile

  v8f acc[4];
#pragma unroll
  for (int nt = 0; nt < 4; ++nt)
#pragma unroll
    for (int r = 0; r < 8; ++r) acc[nt][r] = 0.f;

  for (int kb = 0; kb < Ktot; kb += 32) {
    AB16 a;
#pragma unroll
    for (int j = 0; j < 8; ++j) {
      a.h[j]     = im2col_fetch<KHW, KWW, STRIDE, PAD, Hin, Win>(inimg, kb + koffB + j,      oy, ox);
      a.h[j + 8] = im2col_fetch<KHW, KWW, STRIDE, PAD, Hin, Win>(inimg, kb + koffB + 8 + j,  oy, ox);
    }
#pragma unroll
    for (int nt = 0; nt < 4; ++nt) {
      AB16 b;
      const _Float16* wr = wrow + (size_t)nt * 16 * Ktot;
      b.q[0] = *(const uint4*)(wr + kb + koffB);       // K kb+koffB   .. +7
      b.q[1] = *(const uint4*)(wr + kb + koffB + 8);   // K kb+koffB+8 .. +15
      acc[nt] = __builtin_amdgcn_wmma_f32_16x16x32_f16(
          false, a.v, false, b.v, (short)0, acc[nt], false, false);
    }
  }

  int mbase = mtile * 16 + (lo ? 0 : 8);
#pragma unroll
  for (int nt = 0; nt < 4; ++nt) {
    int   n  = row + 16 * nt;
    float bv = bias[n];
#pragma unroll
    for (int r = 0; r < 8; ++r) {
      float v = acc[nt][r] + bv;
      if (LRELU) v = (v >= 0.f) ? v : NEGS * v;
      int mm  = mbase + r;
      int oy2 = mm / Wout, ox2 = mm % Wout;
      size_t off = (((size_t)img * 64 + n) * Hout + oy2) * Wout + ox2;
      if (outf) outf[off] = v;
      if (outh) outh[off] = (_Float16)v;
    }
  }
}

// ---------------------------------------------------------------------------
// 7. weightnet logits: 1x1-conv MLP (32->32, resblock, ->1), weights in LDS.
//    pool==2 computes AvgPool2 of diff_patch on the fly (LR branch).
// ---------------------------------------------------------------------------
__global__ void __launch_bounds__(256)
weight_logits(const float* __restrict__ dp,   // (5*32, 128, 128)
              const float* __restrict__ w1, const float* __restrict__ b1,
              const float* __restrict__ w2, const float* __restrict__ b2,
              const float* __restrict__ w3, const float* __restrict__ b3,
              const float* __restrict__ w4, const float* __restrict__ b4,
              float* __restrict__ logits, int Hp, int Wp, int pool)
{
  __shared__ float sW1[1024], sW2[1024], sW3[1024];
  __shared__ float sW4[32], sB1[32], sB2[32], sB3[32];
  for (int i = threadIdx.x; i < 1024; i += blockDim.x) {
    sW1[i] = w1[i]; sW2[i] = w2[i]; sW3[i] = w3[i];
  }
  if (threadIdx.x < 32) {
    sW4[threadIdx.x] = w4[threadIdx.x];
    sB1[threadIdx.x] = b1[threadIdx.x];
    sB2[threadIdx.x] = b2[threadIdx.x];
    sB3[threadIdx.x] = b3[threadIdx.x];
  }
  __syncthreads();

  int t  = blockIdx.x * blockDim.x + threadIdx.x;
  int HW = Hp * Wp;
  if (t >= K_NN * HW) return;
  int k   = t / HW;
  int pix = t - k * HW;
  int Yp  = pix / Wp, Xp = pix - Yp * Wp;

  float in[32];
#pragma unroll
  for (int ch = 0; ch < 32; ++ch) {
    const float* pl = dp + (size_t)(k * 32 + ch) * HS * WSZ;
    if (pool == 2) {
      int Y2 = Yp * 2, X2 = Xp * 2;
      in[ch] = 0.25f * (pl[Y2 * WSZ + X2]       + pl[Y2 * WSZ + X2 + 1] +
                        pl[(Y2 + 1) * WSZ + X2] + pl[(Y2 + 1) * WSZ + X2 + 1]);
    } else {
      in[ch] = pl[Yp * WSZ + Xp];
    }
  }

  float h1[32], t2[32];
#pragma unroll
  for (int o = 0; o < 32; ++o) {
    float acc = sB1[o];
#pragma unroll
    for (int i = 0; i < 32; ++i) acc = fmaf(sW1[o * 32 + i], in[i], acc);
    h1[o] = acc;
  }
#pragma unroll
  for (int o = 0; o < 32; ++o) {
    float acc = sB2[o];
#pragma unroll
    for (int i = 0; i < 32; ++i) acc = fmaf(sW2[o * 32 + i], h1[i], acc);
    t2[o] = (acc >= 0.f) ? acc : NEGS * acc;
  }
  float logit = b4[0];
#pragma unroll
  for (int o = 0; o < 32; ++o) {
    float acc = sB3[o];
#pragma unroll
    for (int i = 0; i < 32; ++i) acc = fmaf(sW3[o * 32 + i], t2[i], acc);
    logit = fmaf(sW4[o], h1[o] + acc, logit);   // h = h1 + r; out = W4.h + b4
  }
  logits[t] = logit;
}

// ---------------------------------------------------------------------------
// 8. softmax over K + weighted sum: out[c,pix] = sum_k softmax(logit)_k * vals[k,c,pix]
// ---------------------------------------------------------------------------
__global__ void __launch_bounds__(256)
combine_kernel(const float* __restrict__ vals,   // (5, 64, HW)
               const float* __restrict__ logits, // (5, HW)
               float* __restrict__ out, int HW)
{
  int t = blockIdx.x * blockDim.x + threadIdx.x;
  if (t >= 64 * HW) return;
  int c   = t / HW;
  int pix = t - c * HW;

  float l[K_NN];
  float mx = -3.0e38f;
#pragma unroll
  for (int k = 0; k < K_NN; ++k) { l[k] = logits[k * HW + pix]; mx = fmaxf(mx, l[k]); }
  float se = 0.f;
#pragma unroll
  for (int k = 0; k < K_NN; ++k) { l[k] = __expf(l[k] - mx); se += l[k]; }
  float acc = 0.f;
#pragma unroll
  for (int k = 0; k < K_NN; ++k)
    acc = fmaf(l[k], vals[((size_t)k * 64 + c) * HW + pix], acc);
  out[t] = acc / se;
}

// ---------------------------------------------------------------------------
// Host launcher
// ---------------------------------------------------------------------------
extern "C" void kernel_launch(void* const* d_in, const int* in_sizes, int n_in,
                              void* d_out, int out_size, void* d_ws, size_t ws_size,
                              hipStream_t stream)
{
  const float* y    = (const float*)d_in[0];   // (1,64,64,64)
  const int*   idxk = (const int*)  d_in[1];   // (1,3844,5)
  const float* dp   = (const float*)d_in[2];   // (1,160,128,128)
  const float* k_w1 = (const float*)d_in[3];
  const float* k_b1 = (const float*)d_in[4];
  const float* k_w2 = (const float*)d_in[5];
  const float* k_b2 = (const float*)d_in[6];
  const float* k_w3 = (const float*)d_in[7];
  const float* k_b3 = (const float*)d_in[8];
  const float* l_w1 = (const float*)d_in[9];
  const float* l_b1 = (const float*)d_in[10];
  const float* l_w2 = (const float*)d_in[11];
  const float* l_b2 = (const float*)d_in[12];
  const float* l_w3 = (const float*)d_in[13];
  const float* l_b3 = (const float*)d_in[14];
  const float* l_w4 = (const float*)d_in[15];
  const float* l_b4 = (const float*)d_in[16];
  const float* h_w1 = (const float*)d_in[17];
  const float* h_b1 = (const float*)d_in[18];
  const float* h_w2 = (const float*)d_in[19];
  const float* h_b2 = (const float*)d_in[20];
  const float* h_w3 = (const float*)d_in[21];
  const float* h_b3 = (const float*)d_in[22];
  const float* h_w4 = (const float*)d_in[23];
  const float* h_b4 = (const float*)d_in[24];

  float* out_lr = (float*)d_out;                  // (64,64,64)   = 262144
  float* out_hr = (float*)d_out + 262144;         // (64,128,128) = 1048576

  // ---- workspace suballocation (256B aligned; total ~45 MB) ----
  char* wp = (char*)d_ws;
  auto alloc = [&](size_t bytes) -> char* {
    char* p = wp;
    wp += (bytes + 255) & ~(size_t)255;
    return p;
  };
  float*     scale    = (float*)    alloc((size_t)MPATCH * 64 * 4);
  float*     shiftv   = (float*)    alloc((size_t)MPATCH * 64 * 4);
  float*     khr_f32  = (float*)    alloc((size_t)5 * 64 * HS * WSZ * 4);
  _Float16*  khr_f16  = (_Float16*) alloc((size_t)5 * 64 * HS * WSZ * 2);
  _Float16*  Wh1      = (_Float16*) alloc((size_t)102400 * 2);
  _Float16*  Wh2      = (_Float16*) alloc((size_t)36864 * 2);
  _Float16*  Wh3      = (_Float16*) alloc((size_t)36864 * 2);
  _Float16*  x1h      = (_Float16*) alloc((size_t)5 * 64 * 64 * 64 * 2);
  _Float16*  x2h      = (_Float16*) alloc((size_t)5 * 64 * 64 * 64 * 2);
  float*     klr      = (float*)    alloc((size_t)5 * 64 * 64 * 64 * 4);
  float*     logits_l = (float*)    alloc((size_t)5 * 64 * 64 * 4);
  float*     logits_h = (float*)    alloc((size_t)5 * HS * WSZ * 4);

  // ---- 3. weight conversion ----
  cvt_f32_f16<<<400, 256, 0, stream>>>(k_w1, Wh1, 102400);
  cvt_f32_f16<<<144, 256, 0, stream>>>(k_w2, Wh2, 36864);
  cvt_f32_f16<<<144, 256, 0, stream>>>(k_w3, Wh3, 36864);

  // ---- 1. AdaIN stats (3844*64 threads) ----
  stats_kernel<<<961, 256, 0, stream>>>(y, idxk, scale, shiftv);

  // ---- 2. gather + AdaIN + overlap-average (5*64*128*128 threads) ----
  assemble_hr<<<20480, 256, 0, stream>>>(y, idxk, scale, shiftv, khr_f32, khr_f16);

  // ---- 4-6. convs: 5 imgs x 256 m-tiles = 1280 waves, 4 waves/block ----
  conv_wmma<5, 5, 2, 2, true, HS, WSZ, 64, 64><<<320, 128, 0, stream>>>(
      khr_f16, Wh1, k_b1, (float*)nullptr, x1h);
  conv_wmma<3, 3, 1, 1, true, 64, 64, 64, 64><<<320, 128, 0, stream>>>(
      x1h, Wh2, k_b2, (float*)nullptr, x2h);
  conv_wmma<3, 3, 1, 1, false, 64, 64, 64, 64><<<320, 128, 0, stream>>>(
      x2h, Wh3, k_b3, klr, (_Float16*)nullptr);

  // ---- 7. weightnet logits ----
  weight_logits<<<80, 256, 0, stream>>>(dp, l_w1, l_b1, l_w2, l_b2, l_w3, l_b3,
                                        l_w4, l_b4, logits_l, 64, 64, 2);
  weight_logits<<<320, 256, 0, stream>>>(dp, h_w1, h_b1, h_w2, h_b2, h_w3, h_b3,
                                         h_w4, h_b4, logits_h, HS, WSZ, 1);

  // ---- 8. softmax + blend -> outputs ----
  combine_kernel<<<1024, 256, 0, stream>>>(klr,     logits_l, out_lr, 64 * 64);
  combine_kernel<<<4096, 256, 0, stream>>>(khr_f32, logits_h, out_hr, HS * WSZ);

  (void)in_sizes; (void)n_in; (void)out_size; (void)ws_size;
}
